// NKSSummarySurv_17875653886469
// MI455X (gfx1250) — compile-verified, matrix operand order checked by reading, not run on hardware
//
#include <hip/hip_runtime.h>
#include <hip/hip_bf16.h>

// ---------------------------------------------------------------------------
// NKS summary-survival fused kernel for MI455X (gfx1250, wave32, WMMA).
//
// phi = x@W + b                         (WMMA f16->f32 GEMM, K=256)
// d2  = |phi|^2 + |E|^2 - 2 phi@E^T     (WMMA f16->f32 GEMM, K=128, fused)
// kw  = (d2<=1) ? exp(-d2) : 0          (VALU, kept on-chip — never hits HBM)
// out = clip((kw@KM + g*bKM) / (sum(kw)+g), 1e-12, 1-1e-12)
//                                       (WMMA f16->f32 GEMM, K streamed)
//
// Hot loop: E / KM^T tiles staged Global->LDS with CDNA5 async-DMA
// (global_load_async_to_lds_b128, ASYNCcnt), double-buffered; operands read
// back with bank-conflict-free padded ds_load_b128.
// ---------------------------------------------------------------------------

typedef __attribute__((ext_vector_type(16))) _Float16 v16h;
typedef __attribute__((ext_vector_type(8)))  _Float16 v8h;
typedef __attribute__((ext_vector_type(8)))  float    v8f;

#define B_ROWS 8192
#define D_IN   256
#define D_EMB  128
#define N_EX   16384
#define N_DUR  64
#define N_IT   (N_EX / 32)
#define GAMMA_N 3.6787944117144233e-06f   // exp(-1) / 100000

// workspace layout (bytes, all offsets 256-aligned)
#define OFF_EF   0u          // N_EX*D_EMB  f16  (4 MiB)   exemplar embeddings
#define OFF_EN2  4194304u    // N_EX        f32  (64 KiB)  |E_n|^2
#define OFF_KMT  4259840u    // N_DUR*N_EX  f16  (2 MiB)   exemplar_KM transposed [j][n]
#define OFF_PHI  6356992u    // B*D_EMB     f16  (2 MiB)   encoder output
#define OFF_PN2  8454144u    // B           f32  (32 KiB)  |phi_m|^2
#define OFF_WT   8486912u    // D_EMB*D_IN  f16  (64 KiB)  W transposed [n][k]
#define OFF_GB   8552448u    // N_DUR       f32  (256 B)   gamma * baseline_KM

static __device__ __forceinline__ v16h cat16(v8h a, v8h b) {
  return __builtin_shufflevector(a, b, 0,1,2,3,4,5,6,7,8,9,10,11,12,13,14,15);
}
static __device__ __forceinline__ v8f wmma_f16(v16h a, v16h b, v8f c) {
  // (neg_a, A, neg_b, B, c_mod, C, reuse_a, reuse_b)
  return __builtin_amdgcn_wmma_f32_16x16x32_f16(false, a, false, b, (short)0, c,
                                                false, false);
}
// CDNA5 async DMA: 16B Global -> LDS, tracked by ASYNCcnt (inline asm per
// CDNA5 ISA; low 32 bits of a generic shared pointer are
// the LDS byte offset)
static __device__ __forceinline__ void async_b128(void* lds, const void* g) {
  unsigned off = (unsigned)(size_t)lds;
  asm volatile("global_load_async_to_lds_b128 %0, %1, off"
               :: "v"(off), "v"(g) : "memory");
}

// --- W[k][n] -> Wt[n][k] f16 ------------------------------------------------
__global__ void k_prep_w(const float* __restrict__ W, _Float16* __restrict__ Wt) {
  int idx = blockIdx.x * blockDim.x + threadIdx.x;   // 128*256
  int n = idx >> 8, k = idx & 255;
  Wt[n * D_IN + k] = (_Float16)W[k * D_EMB + n];
}

// --- per-exemplar: E->f16, |E|^2, Kaplan-Meier curve (transposed) -----------
__global__ void k_prep_ex(const float* __restrict__ E,
                          const float* __restrict__ lev,
                          const float* __restrict__ lcen,
                          _Float16* __restrict__ Ef, float* __restrict__ en2,
                          _Float16* __restrict__ KMT) {
  int n = blockIdx.x * blockDim.x + threadIdx.x;
  if (n >= N_EX) return;
  float s = 0.f;
  for (int k = 0; k < D_EMB; ++k) {
    _Float16 h = (_Float16)E[n * D_EMB + k];
    Ef[n * D_EMB + k] = h;
    float f = (float)h;
    s += f * f;
  }
  en2[n] = s;
  // reverse-cumsum at_risk via total-minus-prefix (two passes, no local array)
  float T = 0.f;
  for (int j = 0; j < N_DUR; ++j)
    T += __expf(lev[n * N_DUR + j]) + __expf(lcen[n * N_DUR + j]);
  float ar = T, cum = 0.f;
  for (int j = 0; j < N_DUR; ++j) {
    float e = __expf(lev[n * N_DUR + j]);
    float c = __expf(lcen[n * N_DUR + j]);
    float haz = (ar > 0.f) ? (e / ar) : 0.f;
    cum += __logf(1.f - haz + 1e-7f);
    KMT[j * N_EX + n] = (_Float16)__expf(cum);   // coalesced over n
    ar -= (e + c);
  }
}

// --- gamma * baseline Kaplan-Meier ------------------------------------------
__global__ void k_baseline(const float* __restrict__ logit, float* __restrict__ gb) {
  if (threadIdx.x == 0) {
    float cum = 0.f;
    for (int j = 0; j < N_DUR; ++j) {
      float hz = 1.f / (1.f + __expf(-logit[j]));
      cum += __logf(1.f - hz + 1e-7f);
      gb[j] = GAMMA_N * __expf(cum);
    }
  }
}

// --- encoder: phi = x@W + b, WMMA, 8 waves = 16 rows x 128 cols per block ---
__global__ void k_encoder(const float* __restrict__ x,
                          const _Float16* __restrict__ Wt,
                          const float* __restrict__ bias,
                          _Float16* __restrict__ phi) {
  int lane = threadIdx.x & 31, w = threadIdx.x >> 5;
  int l16 = lane & 15, hb = (lane >> 4) & 1;       // half-wave select
  int rowbase = blockIdx.x * 16;
  int row = rowbase + l16;                         // A-layout: lane -> row M
  int ncol = w * 16 + l16;                         // B-layout: lane -> col N
  v8f acc = {};
#pragma unroll
  for (int kb = 0; kb < 8; ++kb) {                 // K = 256 in chunks of 32
    int K0 = kb * 32 + hb * 8;
    v8f xlo = *(const v8f*)&x[row * D_IN + K0];
    v8f xhi = *(const v8f*)&x[row * D_IN + K0 + 16];
    v16h a = cat16(__builtin_convertvector(xlo, v8h),
                   __builtin_convertvector(xhi, v8h));
    v8h blo = *(const v8h*)&Wt[ncol * D_IN + K0];
    v8h bhi = *(const v8h*)&Wt[ncol * D_IN + K0 + 16];
    acc = wmma_f16(a, cat16(blo, bhi), acc);
  }
  float bv = bias[ncol];
#pragma unroll
  for (int r = 0; r < 8; ++r) {                    // C/D layout: M = r + 8*hb
    int m = rowbase + r + hb * 8;
    phi[m * D_EMB + ncol] = (_Float16)(acc[r] + bv);
  }
}

// --- |phi_m|^2 (from the f16 values actually fed to WMMA) -------------------
__global__ void k_pn2(const _Float16* __restrict__ phi, float* __restrict__ pn2) {
  int m = blockIdx.x * blockDim.x + threadIdx.x;
  if (m >= B_ROWS) return;
  float s = 0.f;
  for (int k = 0; k < D_EMB; k += 8) {
    v8h v = *(const v8h*)&phi[m * D_EMB + k];
#pragma unroll
    for (int i = 0; i < 8; ++i) { float f = (float)v[i]; s += f * f; }
  }
  pn2[m] = s;
}

// --- main fused kernel: 64 blocks x 8 waves; wave = 16 rows, streams N_EX ---
__global__ void __launch_bounds__(256, 2)
k_main(const _Float16* __restrict__ phi, const float* __restrict__ pn2,
       const _Float16* __restrict__ Ef, const float* __restrict__ en2,
       const _Float16* __restrict__ KMT, const float* __restrict__ gb,
       float* __restrict__ out) {
  // double-buffered async-DMA staging tiles; rows padded to dodge bank
  // conflicts on the 16-lane strided B-operand reads
  __shared__ _Float16 Elds[2][32][136];            // E tile:   2 x 8.5 KB
  __shared__ _Float16 Klds[2][64][40];             // KM^T tile: 2 x 5 KB
  __shared__ _Float16 kwlds[8][16][40];            // per-wave kw staging

  int tid = threadIdx.x;
  int lane = tid & 31, w = tid >> 5;
  int l16 = lane & 15, hb = (lane >> 4) & 1;
  int rowbase = blockIdx.x * 128 + w * 16;
  int row = rowbase + l16;

  // async staging assignment: 256 threads move 8KB E + 4KB KM^T per tile
  int erow = tid >> 3, ecol = (tid & 7) * 16;      // 32 rows x 128 halves
  int krow = tid >> 2, kcol = (tid & 3) * 8;       // 64 rows x 32 halves

  // resident A operand: phi rows, K=128 as 4 chunks of 32
  v16h aphi[4];
#pragma unroll
  for (int c = 0; c < 4; ++c) {
    int K0 = c * 32 + hb * 8;
    v8h lo = *(const v8h*)&phi[row * D_EMB + K0];
    v8h hi = *(const v8h*)&phi[row * D_EMB + K0 + 16];
    aphi[c] = cat16(lo, hi);
  }
  float pnv[8];
#pragma unroll
  for (int r = 0; r < 8; ++r) pnv[r] = pn2[rowbase + r + hb * 8];

  v8f acc[4] = {};                                 // 16 rows x 64 durations f32
  float rsum[8] = {};                              // per-row sum(kw) partials

  // prologue: DMA first tile into buffer 0
  {
    const _Float16* eg = &Ef[erow * D_EMB + ecol];
    async_b128(&Elds[0][erow][ecol], eg);
    async_b128(&Elds[0][erow][ecol + 8], eg + 8);
    async_b128(&Klds[0][krow][kcol], &KMT[krow * N_EX + kcol]);
  }

  for (int it = 0; it < N_IT; ++it) {
    int n0 = it * 32, cur = it & 1;
    __syncthreads();                               // readers of buf cur^1 done
    if (it + 1 < N_IT) {                           // DMA next tile into cur^1
      int nn = n0 + 32;
      const _Float16* eg = &Ef[(nn + erow) * D_EMB + ecol];
      async_b128(&Elds[cur ^ 1][erow][ecol], eg);
      async_b128(&Elds[cur ^ 1][erow][ecol + 8], eg + 8);
      async_b128(&Klds[cur ^ 1][krow][kcol], &KMT[krow * N_EX + nn + kcol]);
      if (nn + 32 < N_EX)                          // global_prefetch_b8 ahead
        __builtin_prefetch(&Ef[(nn + 32 + erow) * D_EMB + ecol], 0, 1);
      asm volatile("s_wait_asynccnt 0x3" ::: "memory");  // cur's 3 DMAs done
    } else {
      asm volatile("s_wait_asynccnt 0x0" ::: "memory");
    }
    __syncthreads();                               // cur tile visible to block

#pragma unroll
    for (int nb = 0; nb < 2; ++nb) {               // two 16-exemplar subtiles
      int nl = nb * 16 + l16;
      v8f S = {};
#pragma unroll
      for (int c = 0; c < 4; ++c) {                // phi @ E^T, K=128, from LDS
        int K0 = c * 32 + hb * 8;
        v8h lo = *(const v8h*)&Elds[cur][nl][K0];
        v8h hi = *(const v8h*)&Elds[cur][nl][K0 + 16];
        S = wmma_f16(aphi[c], cat16(lo, hi), S);
      }
      float en = en2[n0 + nl];
#pragma unroll
      for (int r = 0; r < 8; ++r) {
        float d2 = pnv[r] + en - 2.0f * S[r];      // Gram expansion
        d2 = fmaxf(d2, 0.f);
        float kv = (d2 <= 1.0f) ? __expf(-d2) : 0.f;
        rsum[r] += kv;
        kwlds[w][r + hb * 8][nb * 16 + l16] = (_Float16)kv;  // D-layout -> LDS
      }
    }
    // same-wave LDS turnaround: D-layout tile re-read in A-layout
    asm volatile("s_wait_dscnt 0x0" ::: "memory");
    v8h alo = *(const v8h*)&kwlds[w][l16][hb * 8];
    v8h ahi = *(const v8h*)&kwlds[w][l16][hb * 8 + 16];
    v16h akw = cat16(alo, ahi);
#pragma unroll
    for (int jb = 0; jb < 4; ++jb) {               // kw @ KM, 64 durations
      int j = jb * 16 + l16;
      v8h blo = *(const v8h*)&Klds[cur][j][hb * 8];
      v8h bhi = *(const v8h*)&Klds[cur][j][hb * 8 + 16];
      acc[jb] = wmma_f16(akw, cat16(blo, bhi), acc[jb]);
    }
  }

  // butterfly-reduce sum(kw) across the 16 lanes of each half-wave
#pragma unroll
  for (int r = 0; r < 8; ++r) {
    float v = rsum[r];
    v += __shfl_xor(v, 1, 32);
    v += __shfl_xor(v, 2, 32);
    v += __shfl_xor(v, 4, 32);
    v += __shfl_xor(v, 8, 32);
    rsum[r] = v;
  }
#pragma unroll
  for (int jb = 0; jb < 4; ++jb) {
    int j = jb * 16 + l16;
    float g = gb[j];
#pragma unroll
    for (int r = 0; r < 8; ++r) {
      int m = rowbase + r + hb * 8;
      float den = rsum[r] + GAMMA_N + 1e-12f;
      float v = (acc[jb][r] + g) / den;
      v = fminf(fmaxf(v, 1e-12f), 1.0f - 1e-12f);
      out[m * N_DUR + j] = v;
    }
  }
}

extern "C" void kernel_launch(void* const* d_in, const int* in_sizes, int n_in,
                              void* d_out, int out_size, void* d_ws, size_t ws_size,
                              hipStream_t stream) {
  const float* x    = (const float*)d_in[0];
  const float* W    = (const float*)d_in[1];
  const float* bias = (const float*)d_in[2];
  const float* E    = (const float*)d_in[3];
  const float* lev  = (const float*)d_in[4];
  const float* lcen = (const float*)d_in[5];
  const float* lgh  = (const float*)d_in[6];

  char* ws = (char*)d_ws;
  _Float16* Ef  = (_Float16*)(ws + OFF_EF);
  float*    en2 = (float*)(ws + OFF_EN2);
  _Float16* KMT = (_Float16*)(ws + OFF_KMT);
  _Float16* phi = (_Float16*)(ws + OFF_PHI);
  float*    pn2 = (float*)(ws + OFF_PN2);
  _Float16* Wt  = (_Float16*)(ws + OFF_WT);
  float*    gb  = (float*)(ws + OFF_GB);
  float*    out = (float*)d_out;

  k_prep_w  <<<(D_EMB * D_IN) / 256, 256, 0, stream>>>(W, Wt);
  k_prep_ex <<<N_EX / 256, 256, 0, stream>>>(E, lev, lcen, Ef, en2, KMT);
  k_baseline<<<1, 64, 0, stream>>>(lgh, gb);
  k_encoder <<<B_ROWS / 16, 256, 0, stream>>>(x, Wt, bias, phi);
  k_pn2     <<<B_ROWS / 256, 256, 0, stream>>>(phi, pn2);
  k_main    <<<B_ROWS / 128, 256, 0, stream>>>(phi, pn2, Ef, en2, KMT, gb, out);
}
